// Head_11450382812097
// MI455X (gfx1250) — matrix operand order repaired and checked by reading, not compiled, not run on hardware
//
#include <hip/hip_runtime.h>

// ---------------------------------------------------------------------------
// Head attention for B=4, T=2048, C=1024, H=64 on gfx1250 (MI455X).
// All GEMMs via v_wmma_f32_16x16x32_bf16 (bf16 operands, f32 accum).
// ---------------------------------------------------------------------------

typedef __bf16 bf16_t;
typedef __attribute__((ext_vector_type(16))) __bf16 v16bf;
typedef __attribute__((ext_vector_type(8)))  __bf16 v8bf;
typedef __attribute__((ext_vector_type(8)))  float  v8f;
typedef __attribute__((ext_vector_type(4)))  float  v4f;

#define WMMA_BF16(A, B, C) \
    __builtin_amdgcn_wmma_f32_16x16x32_bf16(false, (A), false, (B), (short)0, (C), false, false)

static constexpr int Bb = 4;
static constexpr int T  = 2048;
static constexpr int C  = 1024;
static constexpr int H  = 64;
static constexpr int BT = Bb * T;            // 8192 rows
static constexpr float SCALE = 0.125f;       // H^-0.5

// ---------------------------------------------------------------------------
// Kernel 0: pack Wq/Wk/Wv (f32 [C,64]) -> bf16 B-matrix lane layout.
// Layout: pw[z][t(32)][nt(4)][lane(32)][e(16)]
//   lane<16 : col = nt*16+lane,    K(row) = 32t + e
//   lane>=16: col = nt*16+lane-16, K(row) = 32t + 16 + e
// ---------------------------------------------------------------------------
__global__ void pack_w_kernel(const float* __restrict__ Wq,
                              const float* __restrict__ Wk,
                              const float* __restrict__ Wv,
                              bf16_t* __restrict__ pw) {
    int idx = blockIdx.x * blockDim.x + threadIdx.x;    // 0 .. 12287
    if (idx >= 3 * 32 * 4 * 32) return;
    int lane = idx & 31;
    int nt   = (idx >> 5) & 3;
    int t    = (idx >> 7) & 31;
    int z    = idx >> 12;
    const float* W = (z == 0) ? Wq : (z == 1) ? Wk : Wv;
    int col  = nt * 16 + (lane & 15);
    int krow = t * 32 + ((lane & 16) ? 16 : 0);
    bf16_t* dst = pw + (size_t)idx * 16;
#pragma unroll
    for (int e = 0; e < 16; ++e)
        dst[e] = (bf16_t)W[(size_t)(krow + e) * H + col];
}

// ---------------------------------------------------------------------------
// Kernel 1: q/k/v = x @ W  (f32 out, row-major [BT,64]).
// One wave per (16-row tile, matrix z). 32 K-steps x 4 WMMA.
// B tiles are software-pipelined across K-steps (rotating register sets) so
// the WMMAs depend only on loads issued a full iteration earlier.
// ---------------------------------------------------------------------------
__global__ void qkv_kernel(const float* __restrict__ x,
                           const bf16_t* __restrict__ pw,
                           float* __restrict__ qf,
                           float* __restrict__ kf,
                           float* __restrict__ vf) {
    const int lane = threadIdx.x;                 // 0..31
    const int rt   = blockIdx.x;                  // 0..511
    const int z    = blockIdx.y;                  // 0..2
    const int r0   = rt * 16;
    const bf16_t* pwz = pw + (size_t)z * (32 * 4 * 32 * 16) + lane * 16;
    float* outp = (z == 0) ? qf : (z == 1) ? kf : vf;

    v8f acc[4] = {};
    const float* xrow = x + (size_t)(r0 + (lane & 15)) * C;
    const int koff = (lane & 16) ? 16 : 0;

    // Preload B tiles for t = 0
    v16bf b0 = *(const v16bf*)(pwz);
    v16bf b1 = *(const v16bf*)(pwz + 512);
    v16bf b2 = *(const v16bf*)(pwz + 1024);
    v16bf b3 = *(const v16bf*)(pwz + 1536);

    for (int t = 0; t < 32; ++t) {
        // Prefetch next K-step's B tiles (clamped wrap; overlaps WMMAs)
        const bf16_t* ptn = pwz + (size_t)((t + 1) & 31) * 2048;
        v16bf n0 = *(const v16bf*)(ptn);
        v16bf n1 = *(const v16bf*)(ptn + 512);
        v16bf n2 = *(const v16bf*)(ptn + 1024);
        v16bf n3 = *(const v16bf*)(ptn + 1536);

        // A tile (16x32 bf16) from fp32 x, contiguous k order (k = 16h + e), same as pw
        const int kk = t * 32;
        v4f a0 = *(const v4f*)(xrow + kk + koff);
        v4f a1 = *(const v4f*)(xrow + kk + koff + 4);
        v4f a2 = *(const v4f*)(xrow + kk + koff + 8);
        v4f a3 = *(const v4f*)(xrow + kk + koff + 12);
        v16bf a;
#pragma unroll
        for (int e = 0; e < 4; ++e) {
            a[e]      = (bf16_t)a0[e];
            a[4 + e]  = (bf16_t)a1[e];
            a[8 + e]  = (bf16_t)a2[e];
            a[12 + e] = (bf16_t)a3[e];
        }
        acc[0] = WMMA_BF16(a, b0, acc[0]);
        acc[1] = WMMA_BF16(a, b1, acc[1]);
        acc[2] = WMMA_BF16(a, b2, acc[2]);
        acc[3] = WMMA_BF16(a, b3, acc[3]);

        b0 = n0; b1 = n1; b2 = n2; b3 = n3;
    }

    // C-layout store: VGPR v -> row r0+v (lanes<16) / r0+8+v (lanes>=16)
    const int srow = r0 + ((lane & 16) ? 8 : 0);
    const int scol = lane & 15;
#pragma unroll
    for (int nt = 0; nt < 4; ++nt)
#pragma unroll
        for (int v = 0; v < 8; ++v)
            outp[(size_t)(srow + v) * H + nt * 16 + scol] = acc[nt][v];
}

// ---------------------------------------------------------------------------
// Kernel 2: repack q -> A layout, k -> K^T B layout, v -> V B layout (bf16).
//   qa[qt(512)][s(2)][lane][16] : row = qt*16+(lane&15),
//                                 h = 32s + (lane&16?16:0) + e
//   kb[kt(512)][s(2)][lane][16] : key = kt*16+(lane&15),
//                                 h = 32s + (lane&16?16:0) + e
//   vb[vt(256)][ht(4)][lane][16]: h = ht*16+(lane&15),
//                                 key = vt*32 + (lane&16?16:0) + e
// ---------------------------------------------------------------------------
__global__ void pack_qkv_kernel(const float* __restrict__ qf,
                                const float* __restrict__ kf,
                                const float* __restrict__ vf,
                                bf16_t* __restrict__ qa,
                                bf16_t* __restrict__ kb,
                                bf16_t* __restrict__ vb) {
    int idx = blockIdx.x * blockDim.x + threadIdx.x;   // 0 .. 98303
    if (idx >= 3 * 32768) return;
    int which = idx >> 15;
    int r     = idx & 32767;
    int lane  = r & 31;
    int inner = r >> 5;
    if (which == 0) {
        int qt = inner >> 1, s = inner & 1;
        int row   = qt * 16 + (lane & 15);
        int hbase = 32 * s + ((lane & 16) ? 16 : 0);
        bf16_t* dst = qa + (size_t)r * 16;
#pragma unroll
        for (int e = 0; e < 16; ++e)
            dst[e] = (bf16_t)qf[(size_t)row * H + hbase + e];
    } else if (which == 1) {
        int kt = inner >> 1, s = inner & 1;
        int key   = kt * 16 + (lane & 15);
        int hbase = 32 * s + ((lane & 16) ? 16 : 0);
        bf16_t* dst = kb + (size_t)r * 16;
#pragma unroll
        for (int e = 0; e < 16; ++e)
            dst[e] = (bf16_t)kf[(size_t)key * H + hbase + e];
    } else {
        int vt = inner >> 2, ht = inner & 3;
        int h    = ht * 16 + (lane & 15);
        int kb0  = vt * 32 + ((lane & 16) ? 16 : 0);
        bf16_t* dst = vb + (size_t)r * 16;
#pragma unroll
        for (int e = 0; e < 16; ++e)
            dst[e] = (bf16_t)vf[(size_t)(kb0 + e) * H + h];
    }
}

// ---------------------------------------------------------------------------
// Reductions across one 16-lane half (rows of a C-layout tile).
// ---------------------------------------------------------------------------
__device__ __forceinline__ float half_max16(float x) {
    x = fmaxf(x, __shfl_xor(x, 1, 32));
    x = fmaxf(x, __shfl_xor(x, 2, 32));
    x = fmaxf(x, __shfl_xor(x, 4, 32));
    x = fmaxf(x, __shfl_xor(x, 8, 32));
    return x;
}
__device__ __forceinline__ float half_sum16(float x) {
    x += __shfl_xor(x, 1, 32);
    x += __shfl_xor(x, 2, 32);
    x += __shfl_xor(x, 4, 32);
    x += __shfl_xor(x, 8, 32);
    return x;
}

// ---------------------------------------------------------------------------
// Kernel 3: flash attention, one wave per 16-query tile (512 waves total).
// K tiles are double-buffered (next step's loads issued before the softmax),
// V tiles loaded right after the S WMMAs so both overlap the softmax VALU.
// ---------------------------------------------------------------------------
__global__ void attn_kernel(const bf16_t* __restrict__ qa,
                            const bf16_t* __restrict__ kb,
                            const bf16_t* __restrict__ vb,
                            float* __restrict__ out) {
    __shared__ bf16_t plds[16 * 32];              // per-wave P staging (1 KB)

    const int lane = threadIdx.x;                 // 0..31
    const int qt   = blockIdx.x;                  // 0..511 (global query tile)
    const int b    = qt >> 7;                     // T/16 = 128 tiles / batch
    const int q0   = (qt & 127) << 4;             // query base within batch

    const int rowoff = (lane & 16) ? 8 : 0;
    const int col16  = lane & 15;

    // Q tiles (A layout), h-steps 0 and 1
    const bf16_t* qp = qa + (size_t)qt * 2 * 512 + lane * 16;
    v16bf qa0 = *(const v16bf*)(qp);
    v16bf qa1 = *(const v16bf*)(qp + 512);

    v8f oacc[4] = {};
    float mrun[8], lrun[8];
#pragma unroll
    for (int v = 0; v < 8; ++v) { mrun[v] = -3.402823466e38f; lrun[v] = 0.0f; }

    const float NEG_INF = -__builtin_inff();
    const int nsteps = (q0 + 15) / 32 + 1;

    const bf16_t* kbase = kb + (size_t)(b * 128) * 1024 + lane * 16;
    const bf16_t* vbase = vb + (size_t)(b * 64) * 2048 + lane * 16;

    // Preload K tiles for step 0
    v16bf k00 = *(const v16bf*)(kbase);
    v16bf k01 = *(const v16bf*)(kbase + 512);
    v16bf k10 = *(const v16bf*)(kbase + 1024);
    v16bf k11 = *(const v16bf*)(kbase + 1536);

    for (int j32 = 0; j32 < nsteps; ++j32) {
        const int jbase = j32 * 32;

        // ---- S = Q K^T for 32 keys (two 16-key C tiles) ----
        v8f s0 = {}, s1 = {};
        s0 = WMMA_BF16(qa0, k00, s0);
        s0 = WMMA_BF16(qa1, k01, s0);
        s1 = WMMA_BF16(qa0, k10, s1);
        s1 = WMMA_BF16(qa1, k11, s1);

        // ---- prefetch next step's K tiles (clamped; overlaps softmax) ----
        {
            const int jn = (j32 + 1 < nsteps) ? (j32 + 1) : j32;
            const bf16_t* kn = kbase + (size_t)jn * 2048;
            k00 = *(const v16bf*)(kn);
            k01 = *(const v16bf*)(kn + 512);
            k10 = *(const v16bf*)(kn + 1024);
            k11 = *(const v16bf*)(kn + 1536);
        }
        // ---- V tiles for this step (overlap softmax) ----
        const bf16_t* vp = vbase + (size_t)j32 * 2048;
        v16bf vt0 = *(const v16bf*)(vp);
        v16bf vt1 = *(const v16bf*)(vp + 512);
        v16bf vt2 = *(const v16bf*)(vp + 1024);
        v16bf vt3 = *(const v16bf*)(vp + 1536);

        // ---- scale, causal mask (+ exact-zero quirk), online softmax ----
        float alpha[8];
#pragma unroll
        for (int v = 0; v < 8; ++v) {
            const int qi = q0 + v + rowoff;
            float x0 = s0[v] * SCALE;
            float x1 = s1[v] * SCALE;
            const int k0i = jbase + col16;
            const int k1i = jbase + 16 + col16;
            x0 = (k0i > qi || x0 == 0.0f) ? NEG_INF : x0;
            x1 = (k1i > qi || x1 == 0.0f) ? NEG_INF : x1;
            float mx   = half_max16(fmaxf(x0, x1));
            float mnew = fmaxf(mrun[v], mx);
            float al   = __expf(mrun[v] - mnew);
            float p0   = __expf(x0 - mnew);
            float p1   = __expf(x1 - mnew);
            float rs   = half_sum16(p0 + p1);
            lrun[v] = lrun[v] * al + rs;
            mrun[v] = mnew;
            alpha[v] = al;
            // P -> LDS (C layout scatter), bf16
            plds[(v + rowoff) * 32 + col16]      = (bf16_t)p0;
            plds[(v + rowoff) * 32 + 16 + col16] = (bf16_t)p1;
        }

        asm volatile("s_wait_dscnt 0x0" ::: "memory");

        // ---- P back out in A layout (16 queries x 32 keys) ----
        const bf16_t* pr = plds + (lane & 15) * 32 + ((lane & 16) ? 16 : 0);
        v8bf pl = *(const v8bf*)(pr);
        v8bf ph = *(const v8bf*)(pr + 8);
        v16bf pa;
#pragma unroll
        for (int e = 0; e < 8; ++e) { pa[e] = pl[e]; pa[8 + e] = ph[e]; }

        // ---- O = alpha*O + P V ----
        v8f o0 = oacc[0], o1 = oacc[1], o2 = oacc[2], o3 = oacc[3];
#pragma unroll
        for (int v = 0; v < 8; ++v) {
            o0[v] *= alpha[v]; o1[v] *= alpha[v];
            o2[v] *= alpha[v]; o3[v] *= alpha[v];
        }
        oacc[0] = WMMA_BF16(pa, vt0, o0);
        oacc[1] = WMMA_BF16(pa, vt1, o1);
        oacc[2] = WMMA_BF16(pa, vt2, o2);
        oacc[3] = WMMA_BF16(pa, vt3, o3);
    }

    // ---- normalize and store ----
    const int orow = qt * 16 + rowoff;
#pragma unroll
    for (int nt = 0; nt < 4; ++nt)
#pragma unroll
        for (int v = 0; v < 8; ++v)
            out[(size_t)(orow + v) * H + nt * 16 + col16] = oacc[nt][v] / lrun[v];
}

// ---------------------------------------------------------------------------
// Host launcher. Workspace map (bytes):
//   [0,       393216)  pw  : packed W bf16 (3 x 128 KB)
//   [393216,  2490368) qf  : q fp32 [8192,64]
//   [2490368, 4587520) kf  : k fp32
//   [4587520, 6684672) vf  : v fp32
//   [6684672, 7733248) qa  : packed Q bf16
//   [7733248, 8781824) kb  : packed K^T bf16
//   [8781824, 9830400) vb  : packed V bf16
// ---------------------------------------------------------------------------
extern "C" void kernel_launch(void* const* d_in, const int* in_sizes, int n_in,
                              void* d_out, int out_size, void* d_ws, size_t ws_size,
                              hipStream_t stream) {
    const float* x  = (const float*)d_in[0];
    const float* Wq = (const float*)d_in[1];
    const float* Wk = (const float*)d_in[2];
    const float* Wv = (const float*)d_in[3];
    float* out = (float*)d_out;

    char* ws = (char*)d_ws;
    bf16_t* pw = (bf16_t*)(ws);
    float*  qf = (float*)(ws + 393216);
    float*  kf = (float*)(ws + 2490368);
    float*  vf = (float*)(ws + 4587520);
    bf16_t* qa = (bf16_t*)(ws + 6684672);
    bf16_t* kb = (bf16_t*)(ws + 7733248);
    bf16_t* vb = (bf16_t*)(ws + 8781824);

    pack_w_kernel<<<48, 256, 0, stream>>>(Wq, Wk, Wv, pw);
    qkv_kernel<<<dim3(BT / 16, 3), 32, 0, stream>>>(x, pw, qf, kf, vf);
    pack_qkv_kernel<<<384, 256, 0, stream>>>(qf, kf, vf, qa, kb, vb);
    attn_kernel<<<BT / 16, 32, 0, stream>>>(qa, kb, vb, out);
}
